// InfiniTransformer_40982577938961
// MI455X (gfx1250) — matrix operand order repaired and checked by reading, not compiled, not run on hardware
//
#include <hip/hip_runtime.h>

// ---------------------------------------------------------------------------
// Infini-Transformer layer for MI455X (gfx1250), bf16 WMMA pipeline with
// CDNA5 async global->LDS tile DMA (ASYNCcnt) in the GEMM hot loop.
// B=2, S=4096, D=2048, H=16, DK=DV=128, SEG=512, DH=8192.
// ---------------------------------------------------------------------------

typedef __bf16 bf16_t;
typedef __attribute__((ext_vector_type(16))) __bf16 v16bf;
typedef __attribute__((ext_vector_type(8)))  float  v8f;

union FragAB {
  v16bf v;
  uint4 q[2];
};

__device__ __forceinline__ v8f wmma_bf16f32(const FragAB& a, const FragAB& b, v8f c) {
  // D(16x16,f32) = A(16x32,bf16) x B(32x16,bf16) + C
  return __builtin_amdgcn_wmma_f32_16x16x32_bf16(false, a.v, false, b.v,
                                                 (short)0, c, false, false);
}

__device__ __forceinline__ v8f zero8() {
  v8f z = {0.f, 0.f, 0.f, 0.f, 0.f, 0.f, 0.f, 0.f};
  return z;
}

// CDNA5 async DMA: copy 16 bytes per lane from global memory directly into
// LDS, tracked by ASYNCcnt (CDNA5 ISA). The LDS
// destination VGPR is the low 32 bits of the flat shared address (the LDS
// aperture truncates to addr[31:0], §10.2).
__device__ __forceinline__ void async_ld_b128(unsigned lds_byte_addr, const void* gaddr) {
  asm volatile("global_load_async_to_lds_b128 %0, %1, off"
               :
               : "v"(lds_byte_addr), "v"(gaddr)
               : "memory");
}

constexpr int kB   = 2;
constexpr int kS   = 4096;
constexpr int kD   = 2048;
constexpr int kH   = 16;
constexpr int kDK  = 128;   // == DV
constexpr int kSEG = 512;
constexpr int kNSEG = 8;
constexpr int kDH  = 8192;
constexpr int kM   = kB * kS;  // 8192 rows

// ---------------------------------------------------------------------------
// f32 -> bf16 conversion (straight and transposed for GEMM B operands).
// ---------------------------------------------------------------------------
__global__ __launch_bounds__(256) void cvt_bf16_kernel(const float* __restrict__ in,
                                                       bf16_t* __restrict__ out, int n) {
  for (int i = blockIdx.x * 256 + threadIdx.x; i < n; i += gridDim.x * 256)
    out[i] = (bf16_t)in[i];
}

// W[K][N] f32 -> Wt[N][K] bf16 (so GEMM loads K-contiguous fragments for B).
__global__ __launch_bounds__(256) void cvt_bf16_T_kernel(const float* __restrict__ in,
                                                         bf16_t* __restrict__ out,
                                                         int K, int N) {
  long total = (long)K * N;
  for (long i = (long)blockIdx.x * 256 + threadIdx.x; i < total;
       i += (long)gridDim.x * 256) {
    long k = i / N, n = i % N;
    out[n * (long)K + k] = (bf16_t)in[i];
  }
}

// ---------------------------------------------------------------------------
// Generic bf16 WMMA GEMM:  C[M][N] = act(A[M][K] @ Bt[N][K]^T + bias)
// Block tile 128x128, BK=32, 256 threads = 8 waves, wave tile 64x32.
// Double-buffered LDS fed by async global->LDS DMA; each wave issues 4
// async b128 per stage, so "s_wait_asynccnt 4" retires exactly the current
// stage (async loads complete in order within a wave).
// ---------------------------------------------------------------------------
template <bool OUT_BF16, bool RELU, bool HAS_BIAS>
__global__ __launch_bounds__(256) void gemm_bf16_kernel(
    const bf16_t* __restrict__ A, const bf16_t* __restrict__ Bt,
    const float* __restrict__ bias, void* __restrict__ Cout,
    int M, int N, int K) {
  __shared__ __align__(16) bf16_t sA[2][128 * 32];
  __shared__ __align__(16) bf16_t sB[2][128 * 32];

  const int t = threadIdx.x;
  const int wid = t >> 5, lane = t & 31;
  const int m0 = blockIdx.y * 128, n0 = blockIdx.x * 128;
  const int wm = wid & 1;        // wave row half (64 rows)
  const int wn = wid >> 1;       // wave col quarter (32 cols)

  v8f acc[4][2];
#pragma unroll
  for (int mt = 0; mt < 4; ++mt)
#pragma unroll
    for (int nt = 0; nt < 2; ++nt) acc[mt][nt] = zero8();

  const int lr = t >> 1;            // 0..127 row for cooperative loads
  const int lc = (t & 1) * 16;      // 0 or 16 (16 bf16 per thread)
  const int rsel = lane & 15, kg = lane >> 4;

  const unsigned ldsA[2] = {(unsigned)(size_t)(&sA[0][lr * 32 + lc]),
                            (unsigned)(size_t)(&sA[1][lr * 32 + lc])};
  const unsigned ldsB[2] = {(unsigned)(size_t)(&sB[0][lr * 32 + lc]),
                            (unsigned)(size_t)(&sB[1][lr * 32 + lc])};
  const bf16_t* gA = A + (size_t)(m0 + lr) * K + lc;
  const bf16_t* gB = Bt + (size_t)(n0 + lr) * K + lc;

  auto issue_stage = [&](int stage, int k0) {
    async_ld_b128(ldsA[stage],      gA + k0);
    async_ld_b128(ldsA[stage] + 16, gA + k0 + 8);
    async_ld_b128(ldsB[stage],      gB + k0);
    async_ld_b128(ldsB[stage] + 16, gB + k0 + 8);
  };

  issue_stage(0, 0);  // prologue: stage 0 in flight

  for (int k0 = 0; k0 < K; k0 += 32) {
    const int cur = (k0 >> 5) & 1;
    if (k0 + 32 < K) {
      issue_stage(cur ^ 1, k0 + 32);                    // prefetch next tile
      asm volatile("s_wait_asynccnt 0x4" ::: "memory"); // current tile landed
    } else {
      asm volatile("s_wait_asynccnt 0x0" ::: "memory");
    }
    __syncthreads();

    const bf16_t* cA = sA[cur];
    const bf16_t* cB = sB[cur];

    // ---- fragment loads per ISA 16-bit A/B layouts ----
    FragAB a[4], b[2];
#pragma unroll
    for (int mt = 0; mt < 4; ++mt) {
      const bf16_t* p = cA + (wm * 64 + mt * 16 + rsel) * 32;
      a[mt].q[0] = *(const uint4*)(p + kg * 8);        // K: kg*8 .. +7
      a[mt].q[1] = *(const uint4*)(p + 16 + kg * 8);   // K: 16+kg*8 .. +7
    }
#pragma unroll
    for (int nt = 0; nt < 2; ++nt) {
      const bf16_t* p = cB + (wn * 32 + nt * 16 + rsel) * 32;
      b[nt].q[0] = *(const uint4*)(p + kg * 8);        // K: kg*8 .. +7
      b[nt].q[1] = *(const uint4*)(p + 16 + kg * 8);   // K: 16+kg*8 .. +7
    }

#pragma unroll
    for (int mt = 0; mt < 4; ++mt)
#pragma unroll
      for (int nt = 0; nt < 2; ++nt)
        acc[mt][nt] = wmma_bf16f32(a[mt], b[nt], acc[mt][nt]);
    __syncthreads();  // done reading 'cur' before it gets re-filled next iter
  }

  // ---- epilogue: C layout lane l -> row (l>=16?8:0)+j, col l&15 ----
  const int rhi = lane >> 4;
#pragma unroll
  for (int mt = 0; mt < 4; ++mt) {
#pragma unroll
    for (int nt = 0; nt < 2; ++nt) {
      const int col = n0 + wn * 32 + nt * 16 + rsel;
      const float bv = HAS_BIAS ? bias[col] : 0.f;
#pragma unroll
      for (int j = 0; j < 8; ++j) {
        const int row = m0 + wm * 64 + mt * 16 + rhi * 8 + j;
        float v = acc[mt][nt][j] + bv;
        if (RELU) v = v > 0.f ? v : 0.f;
        if (OUT_BF16)
          ((bf16_t*)Cout)[(size_t)row * N + col] = (bf16_t)v;
        else
          ((float*)Cout)[(size_t)row * N + col] = v;
      }
    }
  }
}

// ---------------------------------------------------------------------------
// Segment-local causal softmax attention (the fully parallel part).
// One workgroup handles one (b, h, seg, 32-row block): two-pass softmax with
// scores resident in LDS, Q@K^T and P@V both via WMMA.
// grid.x = B*H*NSEG*(SEG/32) = 4096.
// Dynamic LDS layout: sQ(8K) sKV(32K) sP(32K) sS(64K) stats(~1.3K) ~= 137.5KB
// ---------------------------------------------------------------------------
__global__ __launch_bounds__(256) void attn_dot_kernel(
    const bf16_t* __restrict__ q_g, const bf16_t* __restrict__ k_g,
    const bf16_t* __restrict__ v_g, bf16_t* __restrict__ dot_g) {
  extern __shared__ __align__(16) char smem[];
  bf16_t* sQ  = (bf16_t*)smem;                       // 32 x 128
  bf16_t* sKV = (bf16_t*)(smem + 8192);              // 128 x 128 (K tile / V^T tile)
  bf16_t* sP  = (bf16_t*)(smem + 8192 + 32768);      // 32 x 512 (exp scores, bf16)
  float*  sS  = (float*)(smem + 8192 + 65536);       // 32 x 512 (raw scores, f32)
  float*  sRed = (float*)(smem + 8192 + 65536 + 65536);  // 32 x 8
  float*  sMax = sRed + 256;                         // 32
  float*  sInv = sMax + 32;                          // 32

  const int t = threadIdx.x, wid = t >> 5, lane = t & 31;
  int bi = blockIdx.x;
  const int rb  = bi & 15;  bi >>= 4;   // 16 row-blocks of 32
  const int seg = bi & 7;   bi >>= 3;
  const int h   = bi & 15;  bi >>= 4;
  const int b   = bi;
  const size_t rowbase = (size_t)b * kS + (size_t)seg * kSEG + rb * 32;
  const size_t segbase = (size_t)b * kS + (size_t)seg * kSEG;
  const int hc = h * 128;

  const int rsel = lane & 15, kg = lane >> 4;
  const int mt  = wid & 1;              // which 16-row tile of the 32-row block
  const int sub = (wid >> 1) * 32;      // 32-col sub-block owned by this wave

  // ---- load Q block: 32 x 128 ----
  {
    const int r = t >> 3;            // 0..31
    const int c = (t & 7) * 16;      // 0..112 step 16
    const uint4* g = (const uint4*)(q_g + (rowbase + r) * kD + hc + c);
    *(uint4*)(sQ + r * 128 + c)     = g[0];
    *(uint4*)(sQ + r * 128 + c + 8) = g[1];
  }

  // ---- phase 1: S = scale * Q @ K^T with causal mask, into sS ----
  const float scale = 0.0883883476483184f;  // 1/sqrt(128)
  for (int kt = 0; kt < 4; ++kt) {
    __syncthreads();  // protect sKV (and makes sQ visible on kt==0)
    {
      const int r = t >> 1;          // key 0..127
      const int cb = (t & 1) * 64;   // dk half
      const uint4* g = (const uint4*)(k_g + (segbase + kt * 128 + r) * kD + hc + cb);
      uint4* d = (uint4*)(sKV + r * 128 + cb);
#pragma unroll
      for (int i = 0; i < 8; ++i) d[i] = g[i];
    }
    __syncthreads();

    v8f s0 = zero8(), s1 = zero8();
#pragma unroll
    for (int ks = 0; ks < 4; ++ks) {
      FragAB a, b0, b1;
      const bf16_t* pa = sQ + (mt * 16 + rsel) * 128 + ks * 32;
      a.q[0] = *(const uint4*)(pa + kg * 8);
      a.q[1] = *(const uint4*)(pa + 16 + kg * 8);
      const bf16_t* pb0 = sKV + (sub + rsel) * 128 + ks * 32;
      b0.q[0] = *(const uint4*)(pb0 + kg * 8);
      b0.q[1] = *(const uint4*)(pb0 + 16 + kg * 8);
      const bf16_t* pb1 = sKV + (sub + 16 + rsel) * 128 + ks * 32;
      b1.q[0] = *(const uint4*)(pb1 + kg * 8);
      b1.q[1] = *(const uint4*)(pb1 + 16 + kg * 8);
      s0 = wmma_bf16f32(a, b0, s0);
      s1 = wmma_bf16f32(a, b1, s1);
    }
    const int rloc = mt * 16 + (lane >> 4) * 8;
#pragma unroll
    for (int j = 0; j < 8; ++j) {
      const int r  = rloc + j;
      const int rg = rb * 32 + r;                 // row within segment
      const int c0 = kt * 128 + sub + rsel;       // key within segment
      const int c1 = c0 + 16;
      sS[r * 512 + c0] = (c0 <= rg) ? s0[j] * scale : -1e30f;
      sS[r * 512 + c1] = (c1 <= rg) ? s1[j] * scale : -1e30f;
    }
  }
  __syncthreads();

  // ---- phase 2: row softmax -> sP (unnormalized exp, bf16), sInv = 1/sum ----
  {
    const int r = t & 31, ch = t >> 5;
    const float* p = sS + r * 512 + ch * 64;
    float mx = -1e30f;
    for (int i = 0; i < 64; ++i) mx = fmaxf(mx, p[i]);
    sRed[r * 8 + ch] = mx;
    __syncthreads();
    if (t < 32) {
      float m = -1e30f;
      for (int i = 0; i < 8; ++i) m = fmaxf(m, sRed[t * 8 + i]);
      sMax[t] = m;
    }
    __syncthreads();
    const float rm = sMax[r];
    bf16_t* pp = sP + r * 512 + ch * 64;
    float sum = 0.f;
    for (int i = 0; i < 64; ++i) {
      const float e = __expf(p[i] - rm);
      pp[i] = (bf16_t)e;
      sum += e;
    }
    sRed[r * 8 + ch] = sum;
    __syncthreads();
    if (t < 32) {
      float s = 0.f;
      for (int i = 0; i < 8; ++i) s += sRed[t * 8 + i];
      sInv[t] = 1.f / s;
    }
  }

  // ---- phase 3: O = P @ V (V transposed into LDS per key tile) ----
  v8f o0 = zero8(), o1 = zero8();
  for (int tt = 0; tt < 4; ++tt) {
    __syncthreads();  // sKV reuse (also covers sP/sInv visibility on tt==0)
    {
      const int r = t >> 1;          // key 0..127
      const int cb = (t & 1) * 64;   // dv half
      union { uint4 q[8]; bf16_t e[64]; } tmp;
      const uint4* g = (const uint4*)(v_g + (segbase + tt * 128 + r) * kD + hc + cb);
#pragma unroll
      for (int i = 0; i < 8; ++i) tmp.q[i] = g[i];
#pragma unroll
      for (int i = 0; i < 64; ++i) sKV[(cb + i) * 128 + r] = tmp.e[i];  // V^T
    }
    __syncthreads();
#pragma unroll
    for (int ks = 0; ks < 4; ++ks) {
      FragAB a, b0, b1;
      const bf16_t* pa = sP + (mt * 16 + rsel) * 512 + tt * 128 + ks * 32;
      a.q[0] = *(const uint4*)(pa + kg * 8);
      a.q[1] = *(const uint4*)(pa + 16 + kg * 8);
      const bf16_t* pb0 = sKV + (sub + rsel) * 128 + ks * 32;
      b0.q[0] = *(const uint4*)(pb0 + kg * 8);
      b0.q[1] = *(const uint4*)(pb0 + 16 + kg * 8);
      const bf16_t* pb1 = sKV + (sub + 16 + rsel) * 128 + ks * 32;
      b1.q[0] = *(const uint4*)(pb1 + kg * 8);
      b1.q[1] = *(const uint4*)(pb1 + 16 + kg * 8);
      o0 = wmma_bf16f32(a, b0, o0);
      o1 = wmma_bf16f32(a, b1, o1);
    }
  }

  // ---- phase 4: normalize rows, store bf16 ----
  const int rloc = mt * 16 + (lane >> 4) * 8;
#pragma unroll
  for (int j = 0; j < 8; ++j) {
    const int r = rloc + j;
    const float inv = sInv[r];
    const size_t base = (rowbase + r) * kD + hc;
    dot_g[base + sub + rsel]      = (bf16_t)(o0[j] * inv);
    dot_g[base + sub + 16 + rsel] = (bf16_t)(o1[j] * inv);
  }
}

// ---------------------------------------------------------------------------
// Sequential Infini-memory recurrence + gate combine (tiny FLOP count).
// One workgroup per (b,h); mem[dk][dv] lives in registers: thread t owns
// dv = t&127 and dk half = t>>7 (64 f32 each). z[128] in LDS.
// Fuses: att = gate*att_mem + (1-gate)*att_dot  -> attb (bf16).
// ---------------------------------------------------------------------------
__global__ __launch_bounds__(256) void attn_mem_kernel(
    const bf16_t* __restrict__ q_g, const bf16_t* __restrict__ k_g,
    const bf16_t* __restrict__ v_g, const bf16_t* __restrict__ dot_g,
    const float* __restrict__ betas, bf16_t* __restrict__ att_g) {
  __shared__ float sSq[128], sSk[128], sV[128], sZ[128], sPart[256], sDen[128];
  const int t = threadIdx.x;
  const int h = blockIdx.x & 15, b = blockIdx.x >> 4;
  const int dv = t & 127, half = t >> 7;
  const int hc = h * 128;

  float m[64];
#pragma unroll
  for (int j = 0; j < 64; ++j) m[j] = 0.f;
  if (t < 128) sZ[t] = 1.0f / 128.0f;  // z0 = 1/DK
  const float gate = 1.f / (1.f + __expf(-betas[hc + dv]));

  for (int seg = 0; seg < kNSEG; ++seg) {
    for (int s = 0; s < kSEG; ++s) {
      const size_t row = (size_t)b * kS + (size_t)seg * kSEG + s;
      __syncthreads();
      if (t < 128) {
        const float qv = (float)q_g[row * kD + hc + t];
        sSq[t] = qv > 0.f ? qv + 1.f : __expf(qv);  // elu(q)+1
      } else {
        const int i = t - 128;
        const float kv = (float)k_g[row * kD + hc + i];
        sSk[i] = kv > 0.f ? kv + 1.f : __expf(kv);  // elu(k)+1
      }
      __syncthreads();
      if (t < 128) {
        sV[t]   = (float)v_g[row * kD + hc + t];
        sDen[t] = sSq[t] * sZ[t];
      }
      // numerator partial: this thread's dk-half for its dv
      float p = 0.f;
#pragma unroll
      for (int j = 0; j < 64; ++j) p += sSq[half * 64 + j] * m[j];
      sPart[t] = p;
      __syncthreads();
      for (int off = 64; off > 0; off >>= 1) {
        if (t < off) sDen[t] += sDen[t + off];
        __syncthreads();
      }
      const float den = sDen[0];
      if (t < 128) {
        const float numer = sPart[t] + sPart[t + 128];
        const float amem  = numer / den;
        const float adot  = (float)dot_g[row * kD + hc + t];
        att_g[row * kD + hc + t] = (bf16_t)(gate * amem + (1.f - gate) * adot);
      }
      // memory update: mem[dk][dv] += sig_k[dk] * v[dv]
      const float vv = sV[dv];
#pragma unroll
      for (int j = 0; j < 64; ++j) m[j] += sSk[half * 64 + j] * vv;
      __syncthreads();
      if (t < 128) sZ[t] += sSk[t];
    }
  }
}

// ---------------------------------------------------------------------------
// Residual + LayerNorm: out = LN(y + x) * g + b   (per 2048-elem row)
// ---------------------------------------------------------------------------
__global__ __launch_bounds__(256) void ln_residual_kernel(
    const float* __restrict__ yf, const float* __restrict__ x,
    const float* __restrict__ g, const float* __restrict__ be,
    float* __restrict__ out) {
  __shared__ float red[256], red2[256];
  const int r = blockIdx.x, t = threadIdx.x;
  const float* py = yf + (size_t)r * kD;
  const float* px = x + (size_t)r * kD;
  float s = 0.f, s2 = 0.f, vals[8];
#pragma unroll
  for (int i = 0; i < 8; ++i) {
    const float v = py[t + i * 256] + px[t + i * 256];
    vals[i] = v;
    s += v;
    s2 += v * v;
  }
  red[t] = s;
  red2[t] = s2;
  __syncthreads();
  for (int off = 128; off > 0; off >>= 1) {
    if (t < off) {
      red[t] += red[t + off];
      red2[t] += red2[t + off];
    }
    __syncthreads();
  }
  const float mu  = red[0] / 2048.f;
  const float var = red2[0] / 2048.f - mu * mu;
  const float inv = rsqrtf(var + 1e-5f);
#pragma unroll
  for (int i = 0; i < 8; ++i) {
    const int c = t + i * 256;
    out[(size_t)r * kD + c] = (vals[i] - mu) * inv * g[c] + be[c];
  }
}

// ---------------------------------------------------------------------------
// Host-side launch sequence.
// ---------------------------------------------------------------------------
extern "C" void kernel_launch(void* const* d_in, const int* in_sizes, int n_in,
                              void* d_out, int out_size, void* d_ws, size_t ws_size,
                              hipStream_t stream) {
  (void)in_sizes; (void)n_in; (void)out_size; (void)ws_size;
  const float* x     = (const float*)d_in[0];
  const float* Wq    = (const float*)d_in[1];
  const float* Wk    = (const float*)d_in[2];
  const float* Wv    = (const float*)d_in[3];
  const float* Wo    = (const float*)d_in[4];
  const float* betas = (const float*)d_in[5];
  const float* W1    = (const float*)d_in[6];
  const float* b1    = (const float*)d_in[7];
  const float* W2    = (const float*)d_in[8];
  const float* b2    = (const float*)d_in[9];
  const float* ln_g  = (const float*)d_in[10];
  const float* ln_b  = (const float*)d_in[11];

  char* ws = (char*)d_ws;
  const size_t MB = 1024ull * 1024ull;
  bf16_t* xb   = (bf16_t*)(ws + 0 * MB);    // 32 MB  x bf16
  bf16_t* qb   = (bf16_t*)(ws + 32 * MB);   // 32 MB
  bf16_t* kbuf = (bf16_t*)(ws + 64 * MB);   // 32 MB
  bf16_t* vbuf = (bf16_t*)(ws + 96 * MB);   // 32 MB
  bf16_t* dotb = (bf16_t*)(ws + 128 * MB);  // 32 MB  softmax-attention output
  bf16_t* attb = (bf16_t*)(ws + 160 * MB);  // 32 MB  gated combined attention
  bf16_t* aob  = (bf16_t*)(ws + 192 * MB);  // 32 MB  attn_out = att @ Wo
  bf16_t* hb   = (bf16_t*)(ws + 224 * MB);  // 128 MB relu(aob@W1+b1)
  float*  yf   = (float*)(ws + 352 * MB);   // 64 MB  h@W2+b2
  bf16_t* WqT  = (bf16_t*)(ws + 416 * MB);  // 8 MB each (transposed bf16)
  bf16_t* WkT  = (bf16_t*)(ws + 424 * MB);
  bf16_t* WvT  = (bf16_t*)(ws + 432 * MB);
  bf16_t* WoT  = (bf16_t*)(ws + 440 * MB);
  bf16_t* W1T  = (bf16_t*)(ws + 448 * MB);  // 32 MB
  bf16_t* W2T  = (bf16_t*)(ws + 480 * MB);  // 32 MB

  dim3 blk(256);

  // precision conversion (+ transpose for B operands)
  cvt_bf16_kernel<<<4096, blk, 0, stream>>>(x, xb, kM * kD);
  cvt_bf16_T_kernel<<<2048, blk, 0, stream>>>(Wq, WqT, kD, kH * kDK);
  cvt_bf16_T_kernel<<<2048, blk, 0, stream>>>(Wk, WkT, kD, kH * kDK);
  cvt_bf16_T_kernel<<<2048, blk, 0, stream>>>(Wv, WvT, kD, kH * kDK);
  cvt_bf16_T_kernel<<<2048, blk, 0, stream>>>(Wo, WoT, kH * kDK, kD);
  cvt_bf16_T_kernel<<<4096, blk, 0, stream>>>(W1, W1T, kD, kDH);
  cvt_bf16_T_kernel<<<4096, blk, 0, stream>>>(W2, W2T, kDH, kD);

  // QKV projections (bf16 out)
  dim3 gQKV(kD / 128, kM / 128);  // (16, 64)
  gemm_bf16_kernel<true, false, false><<<gQKV, blk, 0, stream>>>(xb, WqT, nullptr, qb, kM, kD, kD);
  gemm_bf16_kernel<true, false, false><<<gQKV, blk, 0, stream>>>(xb, WkT, nullptr, kbuf, kM, kD, kD);
  gemm_bf16_kernel<true, false, false><<<gQKV, blk, 0, stream>>>(xb, WvT, nullptr, vbuf, kM, kD, kD);

  // parallel softmax attention (WMMA), then sequential memory + gate combine
  const size_t smem_attn = 8192 + 32768 + 32768 + 65536 + (256 + 32 + 32) * sizeof(float);
  hipFuncSetAttribute((const void*)attn_dot_kernel,
                      hipFuncAttributeMaxDynamicSharedMemorySize, (int)smem_attn);
  attn_dot_kernel<<<kB * kH * kNSEG * (kSEG / 32), blk, smem_attn, stream>>>(qb, kbuf, vbuf, dotb);
  attn_mem_kernel<<<kB * kH, blk, 0, stream>>>(qb, kbuf, vbuf, dotb, betas, attb);

  // output projection + MLP + residual LayerNorm
  gemm_bf16_kernel<true, false, false><<<gQKV, blk, 0, stream>>>(attb, WoT, nullptr, aob, kM, kD, kD);
  dim3 gUP(kDH / 128, kM / 128);  // (64, 64)
  gemm_bf16_kernel<true, true, true><<<gUP, blk, 0, stream>>>(aob, W1T, b1, hb, kM, kDH, kD);
  gemm_bf16_kernel<false, false, true><<<gQKV, blk, 0, stream>>>(hb, W2T, b2, yf, kM, kD, kDH);
  ln_residual_kernel<<<kM, blk, 0, stream>>>(yf, x, ln_g, ln_b, (float*)d_out);
}
